// MultiHeadSelfAttention_4415226380776
// MI455X (gfx1250) — matrix operand order repaired and checked by reading, not compiled, not run on hardware
//
#include <hip/hip_runtime.h>
#include <hip/hip_bf16.h>

typedef __bf16 bf16;
typedef bf16  bf16x8  __attribute__((ext_vector_type(8)));
typedef bf16  bf16x16 __attribute__((ext_vector_type(16)));
typedef float f32x8   __attribute__((ext_vector_type(8)));

#define D_MODEL   1024
#define NUM_HEADS 16
#define D_HEAD    64
#define BATCH     4
#define SEQ       2048
#define M_TOTAL   (BATCH * SEQ)

// LDS row pitches (bf16 elements) chosen for conflict-free 16-lane column reads
#define KPITCH 72   // attention: 32 x 64 K-tile
#define VPITCH 40   // attention: 64 x 32 V-tile
#define PPITCH 40   // attention: 16 x 32 P staging
#define APITCH 40   // gemm: 64 x 32 A-tile
#define BPITCH 40   // gemm: 128 x 32 B-tile

static __device__ __forceinline__ f32x8 zero8() {
  f32x8 z = {0.f, 0.f, 0.f, 0.f, 0.f, 0.f, 0.f, 0.f};
  return z;
}

static __device__ __forceinline__ f32x8 wmma_bf16(bf16x16 a, bf16x16 b, f32x8 c) {
  return __builtin_amdgcn_wmma_f32_16x16x32_bf16(false, a, false, b, (short)0, c,
                                                 false, false);
}

// A fragment (16x32): lane<16 -> row=lane, K {k..k+15}; lane>=16 K {k+16..k+31}
static __device__ __forceinline__ bf16x16 load_fragA(const bf16* __restrict__ base,
                                                     int stride, int row, int kk, int lane) {
  union { bf16x16 v; bf16x8 h[2]; } f;
  const bf16* p = base + (size_t)(row + (lane & 15)) * stride + kk + ((lane >> 4) << 4);
  f.h[0] = *reinterpret_cast<const bf16x8*>(p);
  f.h[1] = *reinterpret_cast<const bf16x8*>(p + 8);
  return f.v;
}

// A fragment from LDS: caller passes pointer at (row, k0); chunks at +0 / +8
static __device__ __forceinline__ bf16x16 load_fragA_lds(const bf16* p) {
  union { bf16x16 v; bf16x8 h[2]; } f;
  f.h[0] = *reinterpret_cast<const bf16x8*>(p);
  f.h[1] = *reinterpret_cast<const bf16x8*>(p + 8);
  return f.v;
}

// B fragment from LDS: 16 contiguous bf16 per lane at caller-computed pointer
static __device__ __forceinline__ bf16x16 load_fragB_lds(const bf16* p) {
  union { bf16x16 v; bf16x8 h[2]; } f;
  f.h[0] = *reinterpret_cast<const bf16x8*>(p);
  f.h[1] = *reinterpret_cast<const bf16x8*>(p + 8);
  return f.v;
}

// CDNA5 async copy: global -> LDS, 16 bytes per lane, tracked by ASYNCcnt
static __device__ __forceinline__ void async_copy_b128(bf16* lds_ptr, const bf16* gptr) {
  uint32_t laddr = (uint32_t)(uintptr_t)lds_ptr;  // flat->LDS: addr[31:0] is the DS address
  asm volatile("global_load_async_to_lds_b128 %0, %1, off"
               :: "v"(laddr), "v"(gptr) : "memory");
}

// ---------------------------------------------------------------- cast f32 -> bf16
__global__ void cast_f32_to_bf16(const float* __restrict__ in, bf16* __restrict__ out, int n) {
  int i = blockIdx.x * blockDim.x + threadIdx.x;
  if (i < n) out[i] = (bf16)in[i];
}

// ---------------------------------------------------------------- V transpose per head
__global__ void transpose_v_kernel(const bf16* __restrict__ V, bf16* __restrict__ Vt) {
  int idx = blockIdx.x * blockDim.x + threadIdx.x;
  if (idx >= BATCH * SEQ * D_MODEL) return;
  int d = idx & (D_MODEL - 1);
  int s = (idx / D_MODEL) & (SEQ - 1);
  int b = idx / (D_MODEL * SEQ);
  int h = d >> 6, dh = d & 63;
  Vt[((size_t)(b * NUM_HEADS + h) * D_HEAD + dh) * SEQ + s] = V[idx];
}

// ---------------------------------------------------------------- mask tile summary
// flags[qt*64+kt] = 1 iff the 16(q) x 32(k) score tile has no masked entries
__global__ void mask_tile_flags(const int* __restrict__ mask, int* __restrict__ flags) {
  int t = blockIdx.x * blockDim.x + threadIdx.x;
  if (t >= (SEQ / 16) * (SEQ / 32)) return;
  int qt = t >> 6, kt = t & 63;
  int all1 = 1;
  const int* row = mask + (size_t)(qt * 16) * SEQ + kt * 32;
  for (int r = 0; r < 16; ++r) {
    for (int c = 0; c < 32; ++c) all1 &= (row[c] != 0);
    row += SEQ;
  }
  flags[t] = all1;
}

// pack per-tile flags into 2 x u32 bitmask words per q-tile (SALU-testable in hot loop)
__global__ void pack_flags(const int* __restrict__ flags, uint32_t* __restrict__ fw) {
  int w = blockIdx.x * blockDim.x + threadIdx.x;  // (SEQ/16)*2 = 256 words
  if (w >= (SEQ / 16) * 2) return;
  uint32_t bits = 0;
  for (int i = 0; i < 32; ++i) bits |= (flags[w * 32 + i] ? 1u : 0u) << i;
  fw[w] = bits;
}

// ---------------------------------------------------------------- GEMM: out = A * W^T + bias
// Double-buffered async-to-LDS pipeline. Block = 8 waves (2x4), wave 32x32, block 64x128.
template <bool OUT_F32>
__global__ __launch_bounds__(256) void gemm_bias_kernel(
    const bf16* __restrict__ A, const bf16* __restrict__ W,
    const float* __restrict__ bias,
    float* __restrict__ outF, bf16* __restrict__ outB,
    int M, int N, int K) {
  __shared__ alignas(16) bf16 lds_a[2][64  * APITCH];
  __shared__ alignas(16) bf16 lds_b[2][128 * BPITCH];

  const int tid   = threadIdx.x;
  const int lane  = tid & 31;
  const int wid   = tid >> 5;
  const int half  = lane >> 4;
  const int cn    = lane & 15;
  const int wm    = wid >> 2, wn = wid & 3;
  const int Abase = blockIdx.y * 64;     // global row of LDS A row 0
  const int Bbase = blockIdx.x * 128;    // global col of LDS B row 0

  const int srow = tid >> 2, sseg = tid & 3;   // staging: 64(A)/128(B) rows x 4 x 16B
  auto stage = [&](int kk, int bufi) {
    async_copy_b128(&lds_a[bufi][srow * APITCH + sseg * 8],
                    A + (size_t)(Abase + srow) * K + kk + sseg * 8);
    async_copy_b128(&lds_b[bufi][srow * BPITCH + sseg * 8],
                    W + (size_t)(Bbase + srow) * K + kk + sseg * 8);
    async_copy_b128(&lds_b[bufi][(64 + srow) * BPITCH + sseg * 8],
                    W + (size_t)(Bbase + 64 + srow) * K + kk + sseg * 8);
  };

  f32x8 c00 = zero8(), c01 = zero8(), c10 = zero8(), c11 = zero8();

  stage(0, 0);
  int buf = 0;
  for (int kk = 0; kk < K; kk += 32) {
    asm volatile("s_wait_asynccnt 0" ::: "memory");
    __syncthreads();
    if (kk + 32 < K) stage(kk + 32, buf ^ 1);
    if (kk + 64 < K) {   // speculative GL2 prefetch two tiles ahead
      __builtin_prefetch(A + (size_t)(Abase + srow) * K + kk + 64, 0, 1);
      __builtin_prefetch(W + (size_t)(Bbase + srow) * K + kk + 64, 0, 1);
    }
    const bf16* ab = &lds_a[buf][0];
    const bf16* bb = &lds_b[buf][0];
    bf16x16 a0 = load_fragA_lds(ab + (wm * 32 + 0  + cn) * APITCH + half * 16);
    bf16x16 a1 = load_fragA_lds(ab + (wm * 32 + 16 + cn) * APITCH + half * 16);
    bf16x16 b0 = load_fragB_lds(bb + (wn * 32 + 0  + cn) * BPITCH + half * 16);
    bf16x16 b1 = load_fragB_lds(bb + (wn * 32 + 16 + cn) * BPITCH + half * 16);
    c00 = wmma_bf16(a0, b0, c00);
    c01 = wmma_bf16(a0, b1, c01);
    c10 = wmma_bf16(a1, b0, c10);
    c11 = wmma_bf16(a1, b1, c11);
    buf ^= 1;
  }

  const int rbase = Abase + wm * 32;
  const int cbase = Bbase + wn * 32;
#pragma unroll
  for (int j = 0; j < 8; ++j) {
    const int r0 = rbase + half * 8 + j;
    const int r1 = r0 + 16;
    const int cA = cbase + cn;
    const int cB = cA + 16;
    float v00 = c00[j] + bias[cA];
    float v01 = c01[j] + bias[cB];
    float v10 = c10[j] + bias[cA];
    float v11 = c11[j] + bias[cB];
    if constexpr (OUT_F32) {
      outF[(size_t)r0 * N + cA] = v00;
      outF[(size_t)r0 * N + cB] = v01;
      outF[(size_t)r1 * N + cA] = v10;
      outF[(size_t)r1 * N + cB] = v11;
    } else {
      outB[(size_t)r0 * N + cA] = (bf16)v00;
      outB[(size_t)r0 * N + cB] = (bf16)v01;
      outB[(size_t)r1 * N + cA] = (bf16)v10;
      outB[(size_t)r1 * N + cB] = (bf16)v11;
    }
  }
}

// ---------------------------------------------------------------- flash attention
__global__ __launch_bounds__(256) void attention_kernel(
    const bf16* __restrict__ Q, const bf16* __restrict__ K,
    const bf16* __restrict__ Vt, const int* __restrict__ mask,
    const uint32_t* __restrict__ fwords, bf16* __restrict__ ctx) {
  __shared__ alignas(16) bf16 lds_k[2][32 * KPITCH];
  __shared__ alignas(16) bf16 lds_v[2][64 * VPITCH];
  __shared__ alignas(16) bf16 lds_p[8][16 * PPITCH];

  const int tid   = threadIdx.x;
  const int lane  = tid & 31;
  const int wid   = tid >> 5;
  const int gw    = blockIdx.x * 8 + wid;
  const int qtile = gw & 127;                 // 8 waves of a block share (b,h)
  const int h     = (gw >> 7) & (NUM_HEADS - 1);
  const int b     = gw >> 11;
  const int qbase = qtile * 16;
  const int half  = lane >> 4;
  const int cn    = lane & 15;

  const bf16* Qb_ = Q  + (size_t)b * SEQ * D_MODEL + h * D_HEAD;
  const bf16* Kb_ = K  + (size_t)b * SEQ * D_MODEL + h * D_HEAD;
  const bf16* Vt_ = Vt + (size_t)(b * NUM_HEADS + h) * D_HEAD * SEQ;

  const int krow = tid >> 3, kseg = tid & 7;
  const int vrow = tid >> 2, vseg = tid & 3;
  auto stage_tiles = [&](int kv, int bufi) {
    async_copy_b128(&lds_k[bufi][krow * KPITCH + kseg * 8],
                    Kb_ + (size_t)(kv + krow) * D_MODEL + kseg * 8);
    async_copy_b128(&lds_v[bufi][vrow * VPITCH + vseg * 8],
                    Vt_ + (size_t)vrow * SEQ + kv + vseg * 8);
  };

  bf16x16 qf0 = load_fragA(Qb_, D_MODEL, qbase, 0,  lane);
  bf16x16 qf1 = load_fragA(Qb_, D_MODEL, qbase, 32, lane);

  // all-ones B fragment: l_acc = P @ Ones gives per-row sums with zero shuffles
  bf16x16 ones;
#pragma unroll
  for (int i = 0; i < 16; ++i) ones[i] = (bf16)1.0f;

  // mask tile bitmasks, hoisted to SGPRs (bit kt == tile fully unmasked)
  const uint32_t fw0 = __builtin_amdgcn_readfirstlane(fwords[qtile * 2 + 0]);
  const uint32_t fw1 = __builtin_amdgcn_readfirstlane(fwords[qtile * 2 + 1]);

  f32x8 o0 = zero8(), o1 = zero8(), o2 = zero8(), o3 = zero8();
  f32x8 lacc = zero8();
  float mrow[8];
#pragma unroll
  for (int j = 0; j < 8; ++j) mrow[j] = -1e30f;
  const float scale = 0.125f;   // 1/sqrt(D_HEAD)

  stage_tiles(0, 0);
  int buf = 0;
  for (int kv = 0; kv < SEQ; kv += 32) {
    asm volatile("s_wait_asynccnt 0" ::: "memory");
    __syncthreads();
    if (kv + 32 < SEQ) stage_tiles(kv + 32, buf ^ 1);

    const bf16* kb = &lds_k[buf][0];
    const bf16* vb = &lds_v[buf][0];

    f32x8 s0 = zero8(), s1 = zero8();
    s0 = wmma_bf16(qf0, load_fragB_lds(kb + (0  + cn) * KPITCH + 0  + half * 16), s0);
    s0 = wmma_bf16(qf1, load_fragB_lds(kb + (0  + cn) * KPITCH + 32 + half * 16), s0);
    s1 = wmma_bf16(qf0, load_fragB_lds(kb + (16 + cn) * KPITCH + 0  + half * 16), s1);
    s1 = wmma_bf16(qf1, load_fragB_lds(kb + (16 + cn) * KPITCH + 32 + half * 16), s1);

    const int kt = kv >> 5;
    const int flag = (int)((((kt & 32) ? fw1 : fw0) >> (kt & 31)) & 1u);

#pragma unroll
    for (int j = 0; j < 8; ++j) {
      const int q = qbase + j + half * 8;
      float a  = s0[j] * scale;
      float bv = s1[j] * scale;
      if (!flag) {   // rare path: per-element mask
        if (mask[q * SEQ + kv + cn] == 0)      a  = -1e30f;
        if (mask[q * SEQ + kv + 16 + cn] == 0) bv = -1e30f;
      }
      float t = fmaxf(a, bv);
      t = fmaxf(t, __shfl_xor(t, 8));
      t = fmaxf(t, __shfl_xor(t, 4));
      t = fmaxf(t, __shfl_xor(t, 2));
      t = fmaxf(t, __shfl_xor(t, 1));
      const float mn  = fmaxf(mrow[j], t);
      const float fac = __expf(mrow[j] - mn);
      const float p0  = __expf(a  - mn);
      const float p1  = __expf(bv - mn);
      mrow[j] = mn;
      o0[j] *= fac; o1[j] *= fac; o2[j] *= fac; o3[j] *= fac;
      lacc[j] *= fac;
      const int prow = j + half * 8;
      lds_p[wid][prow * PPITCH + cn]      = (bf16)p0;
      lds_p[wid][prow * PPITCH + 16 + cn] = (bf16)p1;
    }

    // wave-local LDS RAW: drain ds stores, read P back in A-fragment layout
    asm volatile("s_wait_dscnt 0" ::: "memory");
    union { bf16x16 v; bf16x8 h2[2]; } pf;
    const bf16* lp = &lds_p[wid][cn * PPITCH + (half << 4)];
    pf.h2[0] = *reinterpret_cast<const bf16x8*>(lp);
    pf.h2[1] = *reinterpret_cast<const bf16x8*>(lp + 8);

    o0 = wmma_bf16(pf.v, load_fragB_lds(vb + (0  + cn) * VPITCH + half * 16), o0);
    o1 = wmma_bf16(pf.v, load_fragB_lds(vb + (16 + cn) * VPITCH + half * 16), o1);
    o2 = wmma_bf16(pf.v, load_fragB_lds(vb + (32 + cn) * VPITCH + half * 16), o2);
    o3 = wmma_bf16(pf.v, load_fragB_lds(vb + (48 + cn) * VPITCH + half * 16), o3);
    lacc = wmma_bf16(pf.v, ones, lacc);   // row-sum accumulation, no shuffles

    buf ^= 1;
  }

#pragma unroll
  for (int j = 0; j < 8; ++j) {
    const int q = qbase + j + half * 8;
    const float inv = 1.0f / lacc[j];
    const size_t base = ((size_t)(b * SEQ) + q) * D_MODEL + h * D_HEAD + cn;
    ctx[base]      = (bf16)(o0[j] * inv);
    ctx[base + 16] = (bf16)(o1[j] * inv);
    ctx[base + 32] = (bf16)(o2[j] * inv);
    ctx[base + 48] = (bf16)(o3[j] * inv);
  }
}

// ---------------------------------------------------------------- launch
extern "C" void kernel_launch(void* const* d_in, const int* in_sizes, int n_in,
                              void* d_out, int out_size, void* d_ws, size_t ws_size,
                              hipStream_t stream) {
  const float* x    = (const float*)d_in[0];
  const int*   mask = (const int*)  d_in[1];
  const float* Wq   = (const float*)d_in[2];
  const float* bq   = (const float*)d_in[3];
  const float* Wk   = (const float*)d_in[4];
  const float* bk   = (const float*)d_in[5];
  const float* Wv   = (const float*)d_in[6];
  const float* bv   = (const float*)d_in[7];
  const float* Wo   = (const float*)d_in[8];
  const float* bo   = (const float*)d_in[9];

  char* ws = (char*)d_ws;
  size_t off = 0;
  auto alloc = [&](size_t bytes) {
    char* p = ws + off;
    off += (bytes + 255) & ~(size_t)255;
    return p;
  };

  const int nX = BATCH * SEQ * D_MODEL;
  const int nW = D_MODEL * D_MODEL;
  const int nF = (SEQ / 16) * (SEQ / 32);    // 8192 tile flags
  const int nP = (SEQ / 16) * 2;             // 256 packed words

  bf16* xb   = (bf16*)alloc((size_t)nX * 2);
  bf16* Wqb  = (bf16*)alloc((size_t)nW * 2);
  bf16* Wkb  = (bf16*)alloc((size_t)nW * 2);
  bf16* Wvb  = (bf16*)alloc((size_t)nW * 2);
  bf16* Wob  = (bf16*)alloc((size_t)nW * 2);
  bf16* Qb   = (bf16*)alloc((size_t)nX * 2);
  bf16* Kb   = (bf16*)alloc((size_t)nX * 2);
  bf16* Vb   = (bf16*)alloc((size_t)nX * 2);
  bf16* Vt   = (bf16*)alloc((size_t)nX * 2);
  bf16* ctx  = (bf16*)alloc((size_t)nX * 2);
  int* flags = (int*) alloc((size_t)nF * 4);
  uint32_t* fwords = (uint32_t*)alloc((size_t)nP * 4);

  cast_f32_to_bf16<<<(nX + 255) / 256, 256, 0, stream>>>(x,  xb,  nX);
  cast_f32_to_bf16<<<(nW + 255) / 256, 256, 0, stream>>>(Wq, Wqb, nW);
  cast_f32_to_bf16<<<(nW + 255) / 256, 256, 0, stream>>>(Wk, Wkb, nW);
  cast_f32_to_bf16<<<(nW + 255) / 256, 256, 0, stream>>>(Wv, Wvb, nW);
  cast_f32_to_bf16<<<(nW + 255) / 256, 256, 0, stream>>>(Wo, Wob, nW);
  mask_tile_flags<<<(nF + 255) / 256, 256, 0, stream>>>(mask, flags);
  pack_flags<<<1, 256, 0, stream>>>(flags, fwords);

  dim3 gemm_grid(D_MODEL / 128, M_TOTAL / 64);
  gemm_bias_kernel<false><<<gemm_grid, 256, 0, stream>>>(xb, Wqb, bq, nullptr, Qb,
                                                         M_TOTAL, D_MODEL, D_MODEL);
  gemm_bias_kernel<false><<<gemm_grid, 256, 0, stream>>>(xb, Wkb, bk, nullptr, Kb,
                                                         M_TOTAL, D_MODEL, D_MODEL);
  gemm_bias_kernel<false><<<gemm_grid, 256, 0, stream>>>(xb, Wvb, bv, nullptr, Vb,
                                                         M_TOTAL, D_MODEL, D_MODEL);

  transpose_v_kernel<<<(nX + 255) / 256, 256, 0, stream>>>(Vb, Vt);

  attention_kernel<<<(BATCH * NUM_HEADS * (SEQ / 16)) / 8, 256, 0, stream>>>(
      Qb, Kb, Vt, mask, fwords, ctx);

  gemm_bias_kernel<true><<<gemm_grid, 256, 0, stream>>>(ctx, Wob, bo, (float*)d_out,
                                                        nullptr, M_TOTAL, D_MODEL, D_MODEL);
}
